// MGCN_50087908606117
// MI455X (gfx1250) — compile-verified, matrix operand order checked by reading, not run on hardware
//
#include <hip/hip_runtime.h>
#include <hip/hip_bf16.h>
#include <math.h>

typedef _Float16 h16;
typedef __attribute__((ext_vector_type(16))) _Float16 v16h;
typedef __attribute__((ext_vector_type(8)))  _Float16 v8h;
typedef __attribute__((ext_vector_type(8)))  float    v8f;

#define NUM_ENT 50000
#define NUM_REL 200
#define E_HALF  100000
#define TE      200000
#define DIM     200
#define BSZ     512
#define NF      200
#define FLAT    39200
#define EPSV    1e-5f

// ---------------------------------------------------------------------------
// WMMA GEMM:  C[M,N] = epilogue( A[M,K] @ B[K,N] )
// A: row-major (lda, halves).  Bt: N-major, Bt[n*ldbt + k] = B[k,n].
// Both fragments load as two contiguous v8h (16B) chunks per lane, matching
// the CDNA5 16-bit A-matrix VGPR layout (lanes 0-15: K=8g.., V4-7: K=16+8g..).
// mode 0: C = acc*scale ; mode 1: C = sigmoid(acc + bias[n])
// M%16==0, N%16==0, K%32==0, lda/ldbt %8==0 required.
// ---------------------------------------------------------------------------
__global__ void wmma_gemm_nt(const h16* __restrict__ A, const h16* __restrict__ Bt,
                             float* __restrict__ C,
                             int M, int N, int K, int lda, int ldbt, int ldc,
                             int mode, const float* __restrict__ bias, float scale) {
    int wid  = (blockIdx.x * blockDim.x + threadIdx.x) >> 5;
    int lane = threadIdx.x & 31;
    int ntiles = N >> 4;
    int mt = wid / ntiles;
    int nt = wid - mt * ntiles;
    if (mt >= (M >> 4)) return;                 // wave-uniform: EXEC stays all-1s
    size_t row0 = (size_t)mt << 4, col0 = (size_t)nt << 4;
    int g = lane >> 4, l15 = lane & 15;
    const h16* arow = A  + (row0 + l15) * (size_t)lda;
    const h16* brow = Bt + (col0 + l15) * (size_t)ldbt;
    v8f acc = {};
    for (int k0 = 0; k0 < K; k0 += 32) {
        v8h alo = *(const v8h*)(arow + k0 + 8 * g);
        v8h ahi = *(const v8h*)(arow + k0 + 16 + 8 * g);
        v8h blo = *(const v8h*)(brow + k0 + 8 * g);
        v8h bhi = *(const v8h*)(brow + k0 + 16 + 8 * g);
        v16h a, b;
#pragma unroll
        for (int j = 0; j < 8; ++j) { a[j] = alo[j]; a[j + 8] = ahi[j];
                                      b[j] = blo[j]; b[j + 8] = bhi[j]; }
        acc = __builtin_amdgcn_wmma_f32_16x16x32_f16(
                  false, a, false, b, (short)0, acc, false, false);
    }
    size_t n = col0 + l15;
    size_t mbase = row0 + (size_t)g * 8;
#pragma unroll
    for (int r = 0; r < 8; ++r) {
        float v = acc[r];
        if (mode == 0) v *= scale;
        else           { v += bias[n]; v = 1.0f / (1.0f + __expf(-v)); }
        C[(mbase + r) * (size_t)ldc + n] = v;
    }
}

// ---------------------------------------------------------------------------
// Graph side
// ---------------------------------------------------------------------------
__global__ void deg_kernel(const int* __restrict__ ei, float* __restrict__ deg) {
    int t = blockIdx.x * blockDim.x + threadIdx.x;
    if (t >= TE) return;
    int h = (t < E_HALF) ? 0 : 1;               // row indices: ei[0..TE)
    atomicAdd(&deg[h * NUM_ENT + ei[t]], 1.0f);
}

__global__ void deg_inv_kernel(const float* __restrict__ deg, float* __restrict__ dinv) {
    int t = blockIdx.x * blockDim.x + threadIdx.x;
    if (t >= 2 * NUM_ENT) return;
    float d = deg[t];
    dinv[t] = d > 0.0f ? rsqrtf(d) : 0.0f;
}

// one block per edge; scatter norm*(x[src]*rel*ee) into stacked A (stride 600)
__global__ void edge_scatter(const float* __restrict__ x, const float* __restrict__ rel_emb,
                             const float* __restrict__ edge_embs,
                             const int* __restrict__ ei, const int* __restrict__ et,
                             const float* __restrict__ dinv, float* __restrict__ Aacc) {
    int ge = blockIdx.x;                        // 0..TE
    int d  = threadIdx.x;
    if (d >= DIM) return;
    int h   = (ge < E_HALF) ? 0 : 1;
    int row = ei[ge];
    int col = ei[TE + ge];
    int r   = et[ge];
    const float* dv = dinv + h * NUM_ENT;
    float norm = dv[row] * dv[col];
    float v = x[(size_t)row * DIM + d] * rel_emb[(size_t)r * DIM + d]
            * edge_embs[(size_t)ge * DIM + d] * norm;
    atomicAdd(&Aacc[(size_t)col * 600 + h * DIM + d], v);
}

__global__ void loop_fill(const float* __restrict__ x, const float* __restrict__ loop_rel,
                          const float* __restrict__ loop_edge, float* __restrict__ Aacc) {
    int t = blockIdx.x * blockDim.x + threadIdx.x;
    if (t >= NUM_ENT * DIM) return;
    int d = t % DIM;
    Aacc[(size_t)(t / DIM) * 600 + 400 + d] = x[t] * loop_rel[d] * loop_edge[d];
}

__global__ void a_to_h16(const float* __restrict__ Aacc, h16* __restrict__ Ah) {
    int t = blockIdx.x * blockDim.x + threadIdx.x;
    if (t >= NUM_ENT * 640) return;
    int k = t % 640;
    Ah[t] = (k < 600) ? (h16)Aacc[(size_t)(t / 640) * 600 + k] : (h16)0.0f;
}

__global__ void build_wcat(const float* __restrict__ wi, const float* __restrict__ wo,
                           const float* __restrict__ wl, h16* __restrict__ Wt) {
    int t = blockIdx.x * blockDim.x + threadIdx.x;   // 208 x 640, N-major
    if (t >= 208 * 640) return;
    int f = t / 640, k = t % 640;
    float v = 0.0f;
    if (f < DIM && k < 600) {
        if      (k < 200) v = wi[(size_t)k * DIM + f];
        else if (k < 400) v = wo[(size_t)(k - 200) * DIM + f];
        else              v = wl[(size_t)(k - 400) * DIM + f];
    }
    Wt[t] = (h16)v;
}

// ---------------------------------------------------------------------------
// Stats / BN kernels
// ---------------------------------------------------------------------------
__global__ void col_stats(const float* __restrict__ X, int rows, int ldx,
                          float* __restrict__ mean, float* __restrict__ var) {
    int c = blockIdx.x;
    float s = 0.f, ss = 0.f;
    for (int r = threadIdx.x; r < rows; r += blockDim.x) {
        float v = X[(size_t)r * ldx + c];
        s += v; ss += v * v;
    }
    __shared__ float sh0[256], sh1[256];
    sh0[threadIdx.x] = s; sh1[threadIdx.x] = ss;
    __syncthreads();
    for (int w = 128; w > 0; w >>= 1) {
        if (threadIdx.x < w) { sh0[threadIdx.x] += sh0[threadIdx.x + w];
                               sh1[threadIdx.x] += sh1[threadIdx.x + w]; }
        __syncthreads();
    }
    if (threadIdx.x == 0) {
        float m = sh0[0] / rows;
        mean[c] = m;
        var[c]  = sh1[0] / rows - m * m;
    }
}

__global__ void glob_stats(const float* __restrict__ X, int n, float* __restrict__ mv) {
    float s = 0.f, ss = 0.f;
    for (int t = threadIdx.x; t < n; t += blockDim.x) {
        float v = X[t]; s += v; ss += v * v;
    }
    __shared__ float sh0[256], sh1[256];
    sh0[threadIdx.x] = s; sh1[threadIdx.x] = ss;
    __syncthreads();
    for (int w = 128; w > 0; w >>= 1) {
        if (threadIdx.x < w) { sh0[threadIdx.x] += sh0[threadIdx.x + w];
                               sh1[threadIdx.x] += sh1[threadIdx.x + w]; }
        __syncthreads();
    }
    if (threadIdx.x == 0) {
        float m = sh0[0] / n;
        mv[0] = m; mv[1] = sh1[0] / n - m * m;
    }
}

// tanh(BN(Y)) -> all_ent f32 (gather source) + ent16 N-major padded K=224
__global__ void bn_tanh_emit(const float* __restrict__ Y, const float* __restrict__ mean,
                             const float* __restrict__ var, float* __restrict__ all_ent,
                             h16* __restrict__ ent16) {
    int t = blockIdx.x * blockDim.x + threadIdx.x;
    if (t >= NUM_ENT * 224) return;
    int n = t / 224, k = t % 224;
    h16 hv = (h16)0.0f;
    if (k < DIM) {
        float v = tanhf((Y[(size_t)n * 208 + k] - mean[k]) * rsqrtf(var[k] + EPSV));
        all_ent[(size_t)n * DIM + k] = v;
        hv = (h16)v;
    }
    ent16[t] = hv;
}

// ---------------------------------------------------------------------------
// Decoder side
// ---------------------------------------------------------------------------
__global__ void all_rel_kernel(const float* __restrict__ re, const float* __restrict__ rw,
                               float* __restrict__ all_rel) {
    int r = blockIdx.x, d = threadIdx.x;
    if (d >= DIM) return;
    float s = 0.f;
    for (int k = 0; k < DIM; ++k) s += re[(size_t)r * DIM + k] * rw[(size_t)k * DIM + d];
    all_rel[(size_t)r * DIM + d] = s;
}

__global__ void build_img(const float* __restrict__ all_ent, const float* __restrict__ all_rel,
                          const int* __restrict__ src, const int* __restrict__ rel,
                          float* __restrict__ img) {
    int t = blockIdx.x * blockDim.x + threadIdx.x;  // BSZ*400
    if (t >= BSZ * 400) return;
    int b = t / 400, f = t % 400;
    img[t] = (f & 1) ? all_rel[(size_t)rel[b] * DIM + (f >> 1)]
                     : all_ent[(size_t)src[b] * DIM + (f >> 1)];
}

// im2col with fused input-BN: (BSZ*196) x 64 f16, K padded 49->64
__global__ void im2col_bn(const float* __restrict__ img, const float* __restrict__ mv,
                          h16* __restrict__ col) {
    int t = blockIdx.x * blockDim.x + threadIdx.x;
    if (t >= BSZ * 196 * 64) return;
    int pr = t / 64, c = t % 64;
    h16 hv = (h16)0.0f;
    if (c < 49) {
        int b = pr / 196, p = pr % 196;
        int oy = p / 14, ox = p % 14, ky = c / 7, kx = c % 7;
        float v = (img[(size_t)b * 400 + (oy + ky) * 20 + (ox + kx)] - mv[0])
                  * rsqrtf(mv[1] + EPSV);
        hv = (h16)v;
    }
    col[t] = hv;
}

__global__ void build_wconv(const float* __restrict__ ck, h16* __restrict__ w) {
    int t = blockIdx.x * blockDim.x + threadIdx.x;  // 208 x 64, N-major
    if (t >= 208 * 64) return;
    int f = t / 64, c = t % 64;
    w[t] = (f < NF && c < 49) ? (h16)ck[(size_t)f * 49 + c] : (h16)0.0f;
}

// relu(BN(conv)) reordered into FC-A layout (b, flat = c*196+p), f16
__global__ void build_fcA(const float* __restrict__ co, const float* __restrict__ fm,
                          const float* __restrict__ fv, h16* __restrict__ fcA) {
    long t = (long)blockIdx.x * blockDim.x + threadIdx.x;
    if (t >= (long)BSZ * FLAT) return;
    int b = (int)(t / FLAT), flat = (int)(t % FLAT);
    int c = flat / 196, p = flat % 196;
    float v = co[((size_t)b * 196 + p) * 208 + c];
    v = (v - fm[c]) * rsqrtf(fv[c] + EPSV);
    fcA[t] = (h16)fmaxf(v, 0.0f);
}

__global__ void build_fcw(const float* __restrict__ fw, h16* __restrict__ w16) {
    long t = (long)blockIdx.x * blockDim.x + threadIdx.x;  // 208 x 39200, N-major
    if (t >= (long)208 * FLAT) return;
    int d = (int)(t / FLAT), k = (int)(t % FLAT);
    w16[t] = (d < DIM) ? (h16)fw[(size_t)d * FLAT + k] : (h16)0.0f;
}

// relu(BN(fc)) -> h f16, K padded 200->224 (fc bias cancels under BN axis-0)
__global__ void build_h(const float* __restrict__ fo, const float* __restrict__ hm,
                        const float* __restrict__ hv, h16* __restrict__ hbuf) {
    int t = blockIdx.x * blockDim.x + threadIdx.x;
    if (t >= BSZ * 224) return;
    int b = t / 224, d = t % 224;
    h16 r = (h16)0.0f;
    if (d < DIM) {
        float v = (fo[(size_t)b * 208 + d] - hm[d]) * rsqrtf(hv[d] + EPSV);
        r = (h16)fmaxf(v, 0.0f);
    }
    hbuf[t] = r;
}

// ---------------------------------------------------------------------------
static inline int cdiv(long a, long b) { return (int)((a + b - 1) / b); }

extern "C" void kernel_launch(void* const* d_in, const int* in_sizes, int n_in,
                              void* d_out, int out_size, void* d_ws, size_t ws_size,
                              hipStream_t stream) {
    const float* ent_emb  = (const float*)d_in[0];
    const float* rel_emb  = (const float*)d_in[1];
    const float* edge_emb = (const float*)d_in[2];
    const float* in_w     = (const float*)d_in[3];
    const float* out_w    = (const float*)d_in[4];
    const float* loop_w   = (const float*)d_in[5];
    const float* rels_w   = (const float*)d_in[6];
    const float* loop_rel = (const float*)d_in[7];
    const float* loop_edge= (const float*)d_in[8];
    const float* conv_k   = (const float*)d_in[9];
    const float* fc_w     = (const float*)d_in[10];
    const float* ent_bias = (const float*)d_in[12];
    const int*   src      = (const int*)d_in[13];
    const int*   rel      = (const int*)d_in[14];
    const int*   ei       = (const int*)d_in[16];
    const int*   et       = (const int*)d_in[17];
    float* out = (float*)d_out;

    char* ws = (char*)d_ws;
    size_t off = 0;
    auto alloc = [&](size_t bytes) -> void* {
        void* p = ws + off;
        off = (off + bytes + 255) & ~(size_t)255;
        return p;
    };
    float* deg    = (float*)alloc(2 * NUM_ENT * 4);
    float* dinv   = (float*)alloc(2 * NUM_ENT * 4);
    float* Aacc   = (float*)alloc((size_t)NUM_ENT * 600 * 4);
    h16*   Ah     = (h16*)  alloc((size_t)NUM_ENT * 640 * 2);
    h16*   Wcat   = (h16*)  alloc((size_t)208 * 640 * 2);
    float* Y      = (float*)alloc((size_t)NUM_ENT * 208 * 4);
    float* cm     = (float*)alloc(DIM * 4);
    float* cv     = (float*)alloc(DIM * 4);
    float* allent = (float*)alloc((size_t)NUM_ENT * DIM * 4);
    h16*   ent16  = (h16*)  alloc((size_t)NUM_ENT * 224 * 2);
    float* allrel = (float*)alloc((size_t)2 * NUM_REL * DIM * 4);
    float* img    = (float*)alloc((size_t)BSZ * 400 * 4);
    float* mv0    = (float*)alloc(2 * 4);
    h16*   imcol  = (h16*)  alloc((size_t)BSZ * 196 * 64 * 2);
    h16*   wconv  = (h16*)  alloc((size_t)208 * 64 * 2);
    float* convo  = (float*)alloc((size_t)BSZ * 196 * 208 * 4);
    float* fm     = (float*)alloc(NF * 4);
    float* fv     = (float*)alloc(NF * 4);
    h16*   fcA    = (h16*)  alloc((size_t)BSZ * FLAT * 2);
    h16*   fcw16  = (h16*)  alloc((size_t)208 * FLAT * 2);
    float* fcout  = (float*)alloc((size_t)BSZ * 208 * 4);
    float* hm     = (float*)alloc(DIM * 4);
    float* hvv    = (float*)alloc(DIM * 4);
    h16*   hbuf   = (h16*)  alloc((size_t)BSZ * 224 * 2);

    hipMemsetAsync(deg,  0, 2 * NUM_ENT * 4, stream);
    hipMemsetAsync(Aacc, 0, (size_t)NUM_ENT * 600 * 4, stream);

    // --- graph propagation (scatter, then one stacked 50000x640x208 WMMA GEMM)
    deg_kernel<<<cdiv(TE, 256), 256, 0, stream>>>(ei, deg);
    deg_inv_kernel<<<cdiv(2 * NUM_ENT, 256), 256, 0, stream>>>(deg, dinv);
    edge_scatter<<<TE, 256, 0, stream>>>(ent_emb, rel_emb, edge_emb, ei, et, dinv, Aacc);
    loop_fill<<<cdiv((long)NUM_ENT * DIM, 256), 256, 0, stream>>>(ent_emb, loop_rel, loop_edge, Aacc);
    a_to_h16<<<cdiv((long)NUM_ENT * 640, 256), 256, 0, stream>>>(Aacc, Ah);
    build_wcat<<<cdiv(208 * 640, 256), 256, 0, stream>>>(in_w, out_w, loop_w, Wcat);
    {
        long waves = (NUM_ENT / 16) * (208 / 16);
        wmma_gemm_nt<<<cdiv(waves * 32, 256), 256, 0, stream>>>(
            Ah, Wcat, Y, NUM_ENT, 208, 640, 640, 640, 208, 0, nullptr, 1.0f / 3.0f);
    }
    col_stats<<<DIM, 256, 0, stream>>>(Y, NUM_ENT, 208, cm, cv);
    bn_tanh_emit<<<cdiv((long)NUM_ENT * 224, 256), 256, 0, stream>>>(Y, cm, cv, allent, ent16);

    // --- relations
    all_rel_kernel<<<2 * NUM_REL, 256, 0, stream>>>(rel_emb, rels_w, allrel);

    // --- ConvE decoder
    build_img<<<cdiv(BSZ * 400, 256), 256, 0, stream>>>(allent, allrel, src, rel, img);
    glob_stats<<<1, 256, 0, stream>>>(img, BSZ * 400, mv0);
    im2col_bn<<<cdiv((long)BSZ * 196 * 64, 256), 256, 0, stream>>>(img, mv0, imcol);
    build_wconv<<<cdiv(208 * 64, 256), 256, 0, stream>>>(conv_k, wconv);
    {
        long waves = ((long)BSZ * 196 / 16) * (208 / 16);   // 100352/16 * 13
        wmma_gemm_nt<<<cdiv(waves * 32, 256), 256, 0, stream>>>(
            imcol, wconv, convo, BSZ * 196, 208, 64, 64, 64, 208, 0, nullptr, 1.0f);
    }
    col_stats<<<NF, 256, 0, stream>>>(convo, BSZ * 196, 208, fm, fv);
    build_fcA<<<cdiv((long)BSZ * FLAT, 256), 256, 0, stream>>>(convo, fm, fv, fcA);
    build_fcw<<<cdiv((long)208 * FLAT, 256), 256, 0, stream>>>(fc_w, fcw16);
    {
        long waves = (BSZ / 16) * (208 / 16);
        wmma_gemm_nt<<<cdiv(waves * 32, 256), 256, 0, stream>>>(
            fcA, fcw16, fcout, BSZ, 208, FLAT, FLAT, FLAT, 208, 0, nullptr, 1.0f);
    }
    col_stats<<<DIM, 256, 0, stream>>>(fcout, BSZ, 208, hm, hvv);
    build_h<<<cdiv(BSZ * 224, 256), 256, 0, stream>>>(fcout, hm, hvv, hbuf);

    // --- logits: 512 x 50000, fused +ent_bias and sigmoid, writes d_out directly
    {
        long waves = (BSZ / 16) * (NUM_ENT / 16);
        wmma_gemm_nt<<<cdiv(waves * 32, 256), 256, 0, stream>>>(
            hbuf, ent16, out, BSZ, NUM_ENT, 224, 224, 224, NUM_ENT, 1, ent_bias, 1.0f);
    }
    (void)in_sizes; (void)n_in; (void)out_size; (void)ws_size;
}
